// GBGCN_50818053046586
// MI455X (gfx1250) — matrix operand order repaired and checked by reading, not compiled, not run on hardware
//
#include <hip/hip_runtime.h>
#include <hip/hip_bf16.h>

#define DD 64

typedef __attribute__((ext_vector_type(16))) __bf16 v16bf;
typedef __attribute__((ext_vector_type(8)))  float  v8f;

// f32 -> bf16 bits, round-to-nearest-even (used where we need raw bits)
__device__ __forceinline__ unsigned short bfbits(float f) {
  union { float f; unsigned u; } a; a.f = f;
  unsigned u = a.u + 0x7FFFu + ((a.u >> 16) & 1u);
  return (unsigned short)(u >> 16);
}
__device__ __forceinline__ float bf2f(unsigned short b) {
  union { unsigned u; float f; } c; c.u = ((unsigned)b) << 16;
  return c.f;
}

__device__ __forceinline__ float wave_reduce_add(float v) {
#pragma unroll
  for (int off = 16; off > 0; off >>= 1) v += __shfl_xor(v, off, 32);
  return v;
}

// ---------------- degree count: cnt[dst] += 1 per edge ----------------
__global__ void count_edges(const int* __restrict__ dst, int E, float* __restrict__ cnt) {
  int e = blockIdx.x * blockDim.x + threadIdx.x;
  if (e < E) atomicAdd(&cnt[dst[e]], 1.0f);
}

// -------- one-shot: cnt -> 1/(cnt+1); hoists the IEEE divide out of the GEMM --------
__global__ void finalize_cnt(float* __restrict__ cnt, int n) {
  int i = blockIdx.x * blockDim.x + threadIdx.x;
  if (i < n) cnt[i] = 1.0f / (cnt[i] + 1.0f);
}

// ---------------- f32 -> bf16 bulk convert (weights) ----------------
__global__ void cvt_f32_bf16(const float* __restrict__ in, unsigned short* __restrict__ out, int n) {
  int i = blockIdx.x * blockDim.x + threadIdx.x;
  if (i < n) out[i] = bfbits(in[i]);
}

// ------- scatter: s[dst] += bf16(x[src] * ew) via packed-bf16 atomics -------
// 16 threads/edge, 4 floats each -> 2 pk atomics (2x fewer atomics + bytes than f32)
__global__ void scatter_sum_bf16(const float* __restrict__ x, const int* __restrict__ src,
                                 const int* __restrict__ dst, const float* __restrict__ ew,
                                 unsigned short* __restrict__ s, int E) {
  long t = (long)blockIdx.x * blockDim.x + threadIdx.x;
  long total = (long)E * 16;
  if (t >= total) return;
  int e = (int)(t >> 4);
  int g = (int)(t & 15) * 4;
  int sR = src[e], dR = dst[e];
  float w = ew ? ew[e] : 1.0f;
  const float4 v = *(const float4*)(x + (size_t)sR * DD + g);
  unsigned p0 = (unsigned)bfbits(v.x * w) | ((unsigned)bfbits(v.y * w) << 16);
  unsigned p1 = (unsigned)bfbits(v.z * w) | ((unsigned)bfbits(v.w * w) << 16);
  unsigned long long addr = (unsigned long long)(uintptr_t)(s + (size_t)dR * DD + g);
  asm volatile("global_atomic_pk_add_bf16 %0, %1, off scope:SCOPE_DEV"
               :: "v"(addr), "v"(p0) : "memory");
  asm volatile("global_atomic_pk_add_bf16 %0, %1, off offset:4 scope:SCOPE_DEV"
               :: "v"(addr), "v"(p1) : "memory");
}

// --------- fused GCN dense layer: out = leaky(x@Ws^T + agg@Wn^T + bsum) ---------
// agg = (bf16sum + x) * invcnt (invcnt pre-inverted). One wave = one 16x16 tile,
// 4x v_wmma_f32_16x16x32_bf16. Weights pre-converted to bf16 -> direct v16bf loads.
__global__ __launch_bounds__(256)
void gcn_dense(const float* __restrict__ x, const unsigned short* __restrict__ ssum,
               const float* __restrict__ invcnt,
               const unsigned short* __restrict__ Wsb, const unsigned short* __restrict__ Wnb,
               const float* __restrict__ bs, const float* __restrict__ bn,
               const float* __restrict__ bias,
               float* __restrict__ out, int nrows) {
  const int wave = threadIdx.x >> 5;
  const int lane = threadIdx.x & 31;
  const int col  = (wave & 3) << 4;                      // 4 col tiles of 16
  const int row0 = (blockIdx.x * 2 + (wave >> 2)) << 4;  // 2 row tiles / block
  if (row0 >= nrows) return;                             // wave-uniform; EXEC stays full

  const int m   = lane & 15;
  const int hi  = lane >> 4;
  const int row = row0 + m;
  const int n   = col + m;
  const float inv = invcnt[row];
  const float* xrow = x + (size_t)row * DD;
  const unsigned short* srow = ssum + (size_t)row * DD;

  v8f acc = {};
#pragma unroll
  for (int kb = 0; kb < 64; kb += 32) {
    // B 32x16 layout: lane<16 K=kb..kb+15, lane>=16 K=kb+16..kb+31 -> contiguous 16 bf16
    v16bf b_s = *(const v16bf*)(Wsb + n * DD + kb + hi * 16);
    v16bf b_n = *(const v16bf*)(Wnb + n * DD + kb + hi * 16);
    // A 16x32 layout: lane holds one contiguous run of 16: [base0..+15] (matches B)
    const int base0 = kb + hi * 16;
    uint4 sr0 = *(const uint4*)(srow + base0);        // 8 bf16 = 16B, 16B-aligned
    uint4 sr1 = *(const uint4*)(srow + base0 + 8);
    v16bf a_x, a_g;
#pragma unroll
    for (int e = 0; e < 16; ++e) {
      const int k = (e < 8) ? (base0 + e) : (base0 + 8 + (e - 8));
      const float xf = xrow[k];
      const unsigned pr = (e < 8) ? ((const unsigned*)&sr0)[e >> 1]
                                  : ((const unsigned*)&sr1)[(e - 8) >> 1];
      const unsigned short sb = (e & 1) ? (unsigned short)(pr >> 16) : (unsigned short)(pr & 0xffffu);
      a_x[e] = (__bf16)xf;
      a_g[e] = (__bf16)((bf2f(sb) + xf) * inv);
    }
    acc = __builtin_amdgcn_wmma_f32_16x16x32_bf16(false, a_x, false, b_s, (short)0, acc, false, false);
    acc = __builtin_amdgcn_wmma_f32_16x16x32_bf16(false, a_g, false, b_n, (short)0, acc, false, false);
  }

  const float btot = bs[n] + bn[n] + bias[n];
#pragma unroll
  for (int r = 0; r < 8; ++r) {
    const int mr = r + (hi ? 8 : 0);   // C/D layout: vgpr r -> M = r + 8*hi
    float v = acc[r] + btot;
    v = (v > 0.0f) ? v : 0.2f * v;     // leaky relu
    out[(size_t)(row0 + mr) * DD + n] = v;
  }
}

// --------- fold Wc through Wi/Wp: va,vp (64) and scalar c0 ---------
__global__ void cv_vec(const float* __restrict__ Wi, const float* __restrict__ bi,
                       const float* __restrict__ Wp, const float* __restrict__ bp,
                       const float* __restrict__ Wc, const float* __restrict__ bc,
                       float* __restrict__ va, float* __restrict__ vp, float* __restrict__ c0) {
  int i = threadIdx.x;
  float sa = 0.f, sp = 0.f;
  for (int o = 0; o < 64; ++o) { sa += Wc[o] * Wi[o * 64 + i]; sp += Wc[64 + o] * Wp[o * 64 + i]; }
  va[i] = sa; vp[i] = sp;
  if (i == 0) {
    float c = bc[0];
    for (int o = 0; o < 64; ++o) c += Wc[o] * bi[o] + Wc[64 + o] * bp[o];
    c0[0] = c;
  }
}

// --------- batch stage 1: gather rows, gate, Wt/infl GEMVs, build f=[comb,it,ui,up] ---------
__global__ __launch_bounds__(256)
void batch_front(const int* __restrict__ user_ids, const int* __restrict__ item_ids,
                 const float* __restrict__ init3, const float* __restrict__ part3,
                 const float* __restrict__ soc2, const float* __restrict__ item_emb,
                 const float* __restrict__ va, const float* __restrict__ vpv,
                 const float* __restrict__ c0,
                 const float* __restrict__ Wt, const float* __restrict__ bt,
                 const float* __restrict__ inflW, const float* __restrict__ inflb,
                 float* __restrict__ fbuf, int B) {
  int wave = threadIdx.x >> 5, lane = threadIdx.x & 31;
  int b = blockIdx.x * 8 + wave;
  if (b >= B) return;
  int uid = user_ids[b], iid = item_ids[b];
  const float* ui0 = init3 + (size_t)uid * DD;
  const float* up0 = part3 + (size_t)uid * DD;
  const float* s0  = soc2  + (size_t)uid * DD;

  float p = ui0[2 * lane] * va[2 * lane] + ui0[2 * lane + 1] * va[2 * lane + 1] +
            up0[2 * lane] * vpv[2 * lane] + up0[2 * lane + 1] * vpv[2 * lane + 1];
  p = wave_reduce_add(p);
  float aw = 1.0f / (1.0f + expf(-(p + c0[0])));

  float* f = fbuf + (size_t)b * 256;
#pragma unroll
  for (int j = 0; j < 2; ++j) {
    int o = 2 * lane + j;
    float tp = bt[o], ti = bt[o], sc = inflb[o];
    for (int k = 0; k < DD; ++k) {
      float w = Wt[o * DD + k];
      tp += up0[k] * w;
      ti += ui0[k] * w;
      sc += s0[k] * inflW[o * DD + k];
    }
    float uiv  = ui0[o] + aw * tp;           // init2 row
    float upv  = up0[o] + (1.0f - aw) * ti;  // part2 row
    float comb = 0.6f * uiv + 0.4f * upv + 0.4f * sc;
    f[o]       = comb;
    f[64 + o]  = item_emb[(size_t)iid * DD + o];
    f[128 + o] = uiv;
    f[192 + o] = upv;
  }
}

// --------- batch stage 2: pred MLP (256->128->64->1) and gsp MLP (128->64->1) ---------
__global__ __launch_bounds__(256)
void batch_mlp(const float* __restrict__ fbuf,
               const float* __restrict__ W1, const float* __restrict__ b1,
               const float* __restrict__ W2, const float* __restrict__ b2,
               const float* __restrict__ W3, const float* __restrict__ b3,
               const float* __restrict__ G1, const float* __restrict__ g1b,
               const float* __restrict__ G2, const float* __restrict__ g2b,
               float* __restrict__ out, int B) {
  __shared__ float sh[8][448];   // per wave: f(256) | h1(128) | h2(64)
  int wave = threadIdx.x >> 5, lane = threadIdx.x & 31;
  int b = blockIdx.x * 8 + wave;
  if (b >= B) b = B - 1;         // keep all waves at barriers (B multiple of 8)
  float* f  = sh[wave];
  float* h1 = f + 256;
  float* h2 = f + 384;
#pragma unroll
  for (int j = 0; j < 8; ++j) f[lane + 32 * j] = fbuf[(size_t)b * 256 + lane + 32 * j];
  __syncthreads();
#pragma unroll
  for (int j = 0; j < 4; ++j) {
    int o = lane + 32 * j;
    float acc = b1[o];
    for (int k = 0; k < 256; ++k) acc += f[k] * W1[o * 256 + k];
    h1[o] = fmaxf(acc, 0.0f);
  }
  __syncthreads();
#pragma unroll
  for (int j = 0; j < 2; ++j) {
    int o = lane + 32 * j;
    float acc = b2[o];
    for (int k = 0; k < 128; ++k) acc += h1[k] * W2[o * 128 + k];
    h2[o] = fmaxf(acc, 0.0f);
  }
  __syncthreads();
  float pr = h2[2 * lane] * W3[2 * lane] + h2[2 * lane + 1] * W3[2 * lane + 1];
  pr = wave_reduce_add(pr);
  float rec = 1.0f / (1.0f + expf(-(pr + b3[0])));
  __syncthreads();
#pragma unroll
  for (int j = 0; j < 2; ++j) {   // gsp hidden reuses h1 slot; g == f[:128]
    int o = lane + 32 * j;
    float acc = g1b[o];
    for (int k = 0; k < 128; ++k) acc += f[k] * G1[o * 128 + k];
    h1[o] = fmaxf(acc, 0.0f);
  }
  __syncthreads();
  float ps = h1[2 * lane] * G2[2 * lane] + h1[2 * lane + 1] * G2[2 * lane + 1];
  ps = wave_reduce_add(ps);
  float succ = 1.0f / (1.0f + expf(-(ps + g2b[0])));
  if (lane == 0) { out[b] = rec; out[B + b] = succ; }
}

extern "C" void kernel_launch(void* const* d_in, const int* in_sizes, int n_in,
                              void* d_out, int out_size, void* d_ws, size_t ws_size,
                              hipStream_t stream) {
  (void)n_in; (void)out_size; (void)ws_size;
  const int*   user_ids = (const int*)d_in[0];
  const int*   item_ids = (const int*)d_in[1];
  const int*   ei_init  = (const int*)d_in[2];
  const int*   ei_part  = (const int*)d_in[3];
  const int*   ei_soc   = (const int*)d_in[4];
  const float* soc_w    = (const float*)d_in[5];
  const int B  = in_sizes[0];
  const int Ei = in_sizes[2] / 2;
  const int Ep = in_sizes[3] / 2;
  const int Es = in_sizes[4] / 2;
  // params flattened in dict-insertion order:
  const float* user_emb = (const float*)d_in[6];
  const int U = in_sizes[6] / DD;
  const float* item_emb = (const float*)d_in[7];
  const int INIT0 = 8;   // 3 layers x {Ws,bs,Wn,bn,bias}
  const int PART0 = 23;
  const int SOC0  = 38;  // 2 layers
  const float* inflW = (const float*)d_in[48]; const float* inflb = (const float*)d_in[49];
  const float* Wi = (const float*)d_in[50]; const float* bi = (const float*)d_in[51];
  const float* Wp = (const float*)d_in[52]; const float* bp = (const float*)d_in[53];
  const float* Wc = (const float*)d_in[54]; const float* bc = (const float*)d_in[55];
  const float* Wt = (const float*)d_in[56]; const float* bt = (const float*)d_in[57];
  const float* W1 = (const float*)d_in[58]; const float* b1 = (const float*)d_in[59];
  const float* W2 = (const float*)d_in[60]; const float* b2 = (const float*)d_in[61];
  const float* W3 = (const float*)d_in[62]; const float* b3 = (const float*)d_in[63];
  const float* G1 = (const float*)d_in[64]; const float* g1b = (const float*)d_in[65];
  const float* G2 = (const float*)d_in[66]; const float* g2b = (const float*)d_in[67];

  const size_t UDf = (size_t)U * DD;
  char* p = (char*)d_ws;
  auto alloc = [&](size_t bytes) { void* r = p; p += (bytes + 255) & ~(size_t)255; return r; };
  float*          t0    = (float*)alloc(UDf * 4);      // ping
  float*          t1    = (float*)alloc(UDf * 4);      // pong (also final social emb)
  float*          keepI = (float*)alloc(UDf * 4);      // init_emb after 3 layers
  float*          keepP = (float*)alloc(UDf * 4);      // part_emb after 3 layers
  unsigned short* sbuf  = (unsigned short*)alloc(UDf * 2);  // bf16 scatter accumulator
  unsigned short* wbf   = (unsigned short*)alloc((size_t)16 * 4096 * 2); // bf16 weights
  float*          cntI  = (float*)alloc((size_t)U * 4);
  float*          cntP  = (float*)alloc((size_t)U * 4);
  float*          cntS  = (float*)alloc((size_t)U * 4);
  float*          va    = (float*)alloc(64 * 4);
  float*          vp    = (float*)alloc(64 * 4);
  float*          c0    = (float*)alloc(64 * 4);
  float*          fbuf  = (float*)alloc((size_t)B * 256 * 4);

  // degree counts -> pre-inverted 1/(deg+1)
  hipMemsetAsync(cntI, 0, (size_t)U * 4, stream);
  hipMemsetAsync(cntP, 0, (size_t)U * 4, stream);
  hipMemsetAsync(cntS, 0, (size_t)U * 4, stream);
  count_edges<<<(Ei + 255) / 256, 256, 0, stream>>>(ei_init + Ei, Ei, cntI);
  count_edges<<<(Ep + 255) / 256, 256, 0, stream>>>(ei_part + Ep, Ep, cntP);
  count_edges<<<(Es + 255) / 256, 256, 0, stream>>>(ei_soc + Es, Es, cntS);
  finalize_cnt<<<(U + 255) / 256, 256, 0, stream>>>(cntI, U);
  finalize_cnt<<<(U + 255) / 256, 256, 0, stream>>>(cntP, U);
  finalize_cnt<<<(U + 255) / 256, 256, 0, stream>>>(cntS, U);

  // pre-convert all 8 layers' Ws/Wn to bf16 once
  const int bases[8] = {INIT0, INIT0 + 5, INIT0 + 10, PART0, PART0 + 5, PART0 + 10, SOC0, SOC0 + 5};
  for (int i = 0; i < 8; ++i) {
    cvt_f32_bf16<<<16, 256, 0, stream>>>((const float*)d_in[bases[i]],     wbf + (size_t)(2 * i) * 4096,     4096);
    cvt_f32_bf16<<<16, 256, 0, stream>>>((const float*)d_in[bases[i] + 2], wbf + (size_t)(2 * i + 1) * 4096, 4096);
  }

  auto layer = [&](const float* xin, float* xout, const int* ei, int E,
                   const float* ew, const float* invcnt, int li) {
    hipMemsetAsync(sbuf, 0, UDf * 2, stream);
    long tot = (long)E * 16;
    scatter_sum_bf16<<<(unsigned)((tot + 255) / 256), 256, 0, stream>>>(xin, ei, ei + E, ew, sbuf, E);
    int pb = bases[li];
    gcn_dense<<<(U + 31) / 32, 256, 0, stream>>>(
        xin, sbuf, invcnt,
        wbf + (size_t)(2 * li) * 4096, wbf + (size_t)(2 * li + 1) * 4096,   // Ws, Wn (bf16)
        (const float*)d_in[pb + 1], (const float*)d_in[pb + 3],             // bs, bn
        (const float*)d_in[pb + 4], xout, U);                               // bias
  };

  // initiator view (3 layers)
  layer(user_emb, t0,    ei_init, Ei, nullptr, cntI, 0);
  layer(t0,       t1,    ei_init, Ei, nullptr, cntI, 1);
  layer(t1,       keepI, ei_init, Ei, nullptr, cntI, 2);
  // participant view (3 layers)
  layer(user_emb, t0,    ei_part, Ep, nullptr, cntP, 3);
  layer(t0,       t1,    ei_part, Ep, nullptr, cntP, 4);
  layer(t1,       keepP, ei_part, Ep, nullptr, cntP, 5);
  // social view (2 weighted layers) -> result in t1
  layer(user_emb, t0,    ei_soc, Es, soc_w, cntS, 6);
  layer(t0,       t1,    ei_soc, Es, soc_w, cntS, 7);

  cv_vec<<<1, 64, 0, stream>>>(Wi, bi, Wp, bp, Wc, bc, va, vp, c0);

  batch_front<<<(B + 7) / 8, 256, 0, stream>>>(user_ids, item_ids, keepI, keepP, t1,
                                               item_emb, va, vp, c0, Wt, bt, inflW, inflb,
                                               fbuf, B);
  batch_mlp<<<(B + 7) / 8, 256, 0, stream>>>(fbuf, W1, b1, W2, b2, W3, b3,
                                             G1, g1b, G2, g2b, (float*)d_out, B);
}
